// Intra_channel_27084063769111
// MI455X (gfx1250) — matrix operand order repaired and checked by reading, not compiled, not run on hardware
//
#include <hip/hip_runtime.h>
#include <cstdint>

// ---------------------------------------------------------------------------
// Fused non-local attention block for gfx1250 (MI455X), two-phase pipeline:
//   Phase P: elementwise prep -> theta/phi/g as bf16 pairs in WMMA-fragment-
//            swizzled global layouts (in d_ws).
//   Phase A: flash-attention with double-buffered ASYNC global->LDS tile DMA
//            (global_load_async_to_lds_b128 / s_wait_asynccnt) feeding
//            v_wmma_f32_16x16x32_bf16, 2-deep pipelined LDS fragment loads,
//            ds_swizzle-based softmax reductions.
//   Fallback fused kernel if ws too small.
// ---------------------------------------------------------------------------

typedef __attribute__((ext_vector_type(16))) __bf16       v16bf;
typedef __attribute__((ext_vector_type(8)))  float        v8f;
typedef __attribute__((ext_vector_type(4)))  unsigned int v4u;

#define C_DIM 256
#define HW_DIM 2304
#define QT 64               // query rows per block
#define KT 64               // keys per k-block
#define NKB (HW_DIM / KT)   // 36
#define RSTR 20             // LDS row stride in dwords (mult of 4, conflict-free)

#define WS_U32_PER_ARR 2359296           // 8*8*2304*16 == 8*72*256*16
#define WS_BYTES_NEEDED (3u * WS_U32_PER_ARR * 4u)

__device__ __forceinline__ uint32_t bfr_bits(float f) {
    uint32_t u = __builtin_bit_cast(uint32_t, f);
    u += 0x7fffu + ((u >> 16) & 1u);   // round-to-nearest-even to bf16
    return u >> 16;
}
__device__ __forceinline__ uint32_t pack2bf(float lo, float hi) {
    return bfr_bits(lo) | (bfr_bits(hi) << 16);
}

struct frag32 { v4u lo; v4u hi; };      // 32B == v16bf

__device__ __forceinline__ v16bf load_frag(const uint32_t* p0, const uint32_t* p1) {
    frag32 f;
    f.lo = *(const v4u*)p0;
    f.hi = *(const v4u*)p1;
    return __builtin_bit_cast(v16bf, f);
}

// xor-shuffle within 16-lane groups via ds_swizzle (masks < 16 never cross)
template <int MASK>
__device__ __forceinline__ float xor_sw(float v) {
    return __builtin_bit_cast(float, __builtin_amdgcn_ds_swizzle(
        __builtin_bit_cast(int, v), 0x1f | (MASK << 10)));
}

// ---- CDNA5 async global->LDS copy (ASYNCcnt) ------------------------------
__device__ __forceinline__ void async_b128(uint32_t lds_byte_off, uint64_t gaddr) {
    asm volatile("global_load_async_to_lds_b128 %0, %1, off"
                 :: "v"(lds_byte_off), "v"(gaddr) : "memory");
}
__device__ __forceinline__ void wait_async0() {
    asm volatile("s_wait_asynccnt 0x0" ::: "memory");
}

// ===========================================================================
// Phase P1: theta & phi  -> bf16 C-pairs, layout [b][kc(8)][pos(2304)][pp(16)]
// ===========================================================================
__global__ __launch_bounds__(256)
void prep_theta_phi_kernel(const float* __restrict__ x,
                           const float* __restrict__ w_th, const float* __restrict__ b_th,
                           const float* __restrict__ w_ph, const float* __restrict__ b_ph,
                           uint32_t* __restrict__ Tg, uint32_t* __restrict__ Pg)
{
    int flat = blockIdx.x * 256 + threadIdx.x;          // [0, 8*128*2304)
    int n  = flat % HW_DIM;
    int t  = flat / HW_DIM;
    int cp = t % 128;
    int b  = t / 128;
    int c0 = 2 * cp;
    const float* xb = x + (size_t)b * C_DIM * HW_DIM;
    float x0 = xb[(size_t)c0 * HW_DIM + n];
    float x1 = xb[(size_t)(c0 + 1) * HW_DIM + n];
    size_t idx = ((size_t)(b * 8 + (cp >> 4)) * HW_DIM + n) * 16 + (cp & 15);
    Tg[idx] = pack2bf(x0 * w_th[c0] + b_th[c0], x1 * w_th[c0 + 1] + b_th[c0 + 1]);
    Pg[idx] = pack2bf(x0 * w_ph[c0] + b_ph[c0], x1 * w_ph[c0 + 1] + b_ph[c0 + 1]);
}

// ===========================================================================
// Phase P2: g -> bf16 key-pairs, layout [b][kpg(72)][c(256)][sv(16)]
// ===========================================================================
__global__ __launch_bounds__(256)
void prep_g_kernel(const float* __restrict__ x,
                   const float* __restrict__ w_g, const float* __restrict__ b_g,
                   uint32_t* __restrict__ Vg)
{
    int flat = blockIdx.x * 256 + threadIdx.x;          // [0, 8*256*1152)
    int kp = flat % 1152;
    int t  = flat / 1152;
    int c  = t % C_DIM;
    int b  = t / C_DIM;
    const float* xr = x + ((size_t)b * C_DIM + c) * HW_DIM + 2 * kp;
    float wc = w_g[c], bc = b_g[c];
    size_t idx = ((size_t)(b * 72 + (kp >> 4)) * C_DIM + c) * 16 + (kp & 15);
    Vg[idx] = pack2bf(xr[0] * wc + bc, xr[1] * wc + bc);
}

// ===========================================================================
// Phase A: flash attention with async double-buffered K/V tiles.
// LDS (dwords): sKf[2][10240], sVf[2][10240], sPf 2560, comb 512 -> 176128 B
// ===========================================================================
#define SMEM_ASYNC ((2 * 10240 + 2 * 10240 + 2560 + 512) * 4)
#define KBUF_DW 10240
#define VBUF_DW 10240

__device__ __forceinline__ void issue_tile(int b, int tid, int kb,
                                           uint32_t ldsK, uint32_t ldsV,
                                           const uint32_t* Pg, const uint32_t* Vg)
{
    const uint32_t q16 = (uint32_t)(tid & 3) * 16;   // byte offset within 16-dw row
    const int      t4  = tid >> 2;
    const uint64_t gK0 = (uint64_t)(uintptr_t)Pg + (uint64_t)kb * 4096u;
    const uint64_t gV0 = (uint64_t)(uintptr_t)Vg + (uint64_t)kb * 32768u;
    #pragma unroll
    for (int i = 0; i < 16; ++i) {                   // K tile: 8 kc x 64 keys
        int kc  = i >> 1;
        int key = (i & 1) * 32 + t4;
        uint64_t g = gK0 + (uint64_t)((b * 8 + kc) * HW_DIM + key) * 64u + q16;
        uint32_t l = ldsK + (uint32_t)((kc * 64 + key) * (RSTR * 4)) + q16;
        async_b128(l, g);
    }
    #pragma unroll
    for (int i = 0; i < 16; ++i) {                   // V tile: 2 kc2 x 256 ch
        int kc2 = i >> 3;
        int c   = (i & 7) * 32 + t4;
        uint64_t g = gV0 + (uint64_t)((b * 72 + kc2) * C_DIM + c) * 64u + q16;
        uint32_t l = ldsV + (uint32_t)((kc2 * 256 + c) * (RSTR * 4)) + q16;
        async_b128(l, g);
    }
}

__global__ __launch_bounds__(128)
void nl_attn_async_kernel(const float* __restrict__ x,
                          const uint32_t* __restrict__ Tg,
                          const uint32_t* __restrict__ Pg,
                          const uint32_t* __restrict__ Vg,
                          const float* __restrict__ w_o,  const float* __restrict__ b_o,
                          const float* __restrict__ bn_g, const float* __restrict__ bn_b,
                          const float* __restrict__ bn_m, const float* __restrict__ bn_v,
                          float* __restrict__ out)
{
    extern __shared__ unsigned char smem_raw[];
    uint32_t* sKf  = (uint32_t*)smem_raw;            // [2][10240]
    uint32_t* sVf  = sKf + 2 * KBUF_DW;              // [2][10240]
    uint32_t* sPf  = sVf + 2 * VBUF_DW;              // 2560
    float*    combS = (float*)(sPf + 2 * 64 * RSTR);
    float*    combB = combS + 256;

    const int b    = blockIdx.y;
    const int q0   = blockIdx.x * QT;
    const int tid  = threadIdx.x;
    const int lane = tid & 31;
    const int wave = tid >> 5;
    const int hlf  = lane >> 4;
    const int ln   = lane & 15;
    const float* xb = x + (size_t)b * C_DIM * HW_DIM;

    const uint32_t ldsK0 = (uint32_t)(uintptr_t)sKf;
    const uint32_t ldsV0 = (uint32_t)(uintptr_t)sVf;

    // kick off DMA of first K/V tile immediately
    issue_tile(b, tid, 0, ldsK0, ldsV0, Pg, Vg);

    // fused epilogue params
    for (int c = tid; c < C_DIM; c += 128) {
        float iv = bn_g[c] * rsqrtf(bn_v[c] + 1e-5f);
        combS[c] = w_o[c] * iv;
        combB[c] = b_o[c] * iv + bn_b[c] - bn_m[c] * iv;
    }

    // Q A-fragments straight from global (swizzled) into registers
    const int qrow = wave * 16 + ln;
    v16bf aQ[8];
    #pragma unroll
    for (int kc = 0; kc < 8; ++kc) {
        const uint32_t* base = Tg + ((size_t)(b * 8 + kc) * HW_DIM + q0 + qrow) * 16 + 4 * hlf;
        aQ[kc] = load_frag(base, base + 8);
    }

    v8f accO[16];
    {
        v8f z = {0.f, 0.f, 0.f, 0.f, 0.f, 0.f, 0.f, 0.f};
        #pragma unroll
        for (int i = 0; i < 16; ++i) accO[i] = z;
    }
    float m_r[8], l_r[8];
    #pragma unroll
    for (int j = 0; j < 8; ++j) { m_r[j] = -1e30f; l_r[j] = 0.f; }

    wait_async0();
    __syncthreads();          // first tile resident

    for (int kb = 0; kb < NKB; ++kb) {
        const int cur = kb & 1;
        const uint32_t* sK = sKf + cur * KBUF_DW;
        const uint32_t* sV = sVf + cur * VBUF_DW;

        if (kb + 1 < NKB)     // DMA next tile into the other buffer (overlaps GEMMs)
            issue_tile(b, tid, kb + 1,
                       ldsK0 + (1 - cur) * (KBUF_DW * 4),
                       ldsV0 + (1 - cur) * (VBUF_DW * 4), Pg, Vg);

        // ---- S = Q(16x256) x K^T(256x64), 2-deep pipelined B fragments ----
        v8f accS[4];
        {
            v8f z = {0.f, 0.f, 0.f, 0.f, 0.f, 0.f, 0.f, 0.f};
            #pragma unroll
            for (int i = 0; i < 4; ++i) accS[i] = z;
        }
        {
            auto loadB = [&](int s) -> v16bf {
                int kc = s >> 2, nt = s & 3;
                const uint32_t* bb = sK + ((kc * 64) + nt * 16 + ln) * RSTR + 4 * hlf;
                return load_frag(bb, bb + 8);
            };
            v16bf bp[2];
            bp[0] = loadB(0);
            bp[1] = loadB(1);
            #pragma unroll
            for (int s = 0; s < 32; ++s) {
                v16bf curf = bp[s & 1];
                if (s + 2 < 32) bp[s & 1] = loadB(s + 2);
                accS[s & 3] = __builtin_amdgcn_wmma_f32_16x16x32_bf16(
                    false, aQ[s >> 2], false, curf, (short)0, accS[s & 3], false, false);
            }
        }

        // ---- online softmax (rows = j + 8*hlf; reductions via ds_swizzle) ----
        #pragma unroll
        for (int j = 0; j < 8; ++j) {
            float mv = fmaxf(fmaxf(accS[0][j], accS[1][j]),
                             fmaxf(accS[2][j], accS[3][j]));
            mv = fmaxf(mv, xor_sw<1>(mv));
            mv = fmaxf(mv, xor_sw<2>(mv));
            mv = fmaxf(mv, xor_sw<4>(mv));
            mv = fmaxf(mv, xor_sw<8>(mv));
            float mn = fmaxf(m_r[j], mv);
            float al = __expf(m_r[j] - mn);
            m_r[j] = mn;
            float rs = 0.f;
            #pragma unroll
            for (int nt = 0; nt < 4; ++nt) {
                float p = __expf(accS[nt][j] - mn);
                accS[nt][j] = p;
                rs += p;
            }
            rs += xor_sw<1>(rs);
            rs += xor_sw<2>(rs);
            rs += xor_sw<4>(rs);
            rs += xor_sw<8>(rs);
            l_r[j] = l_r[j] * al + rs;
            #pragma unroll
            for (int ct = 0; ct < 16; ++ct) accO[ct][j] *= al;
        }

        // ---- P -> LDS, A-fragment swizzled ----
        {
            uint16_t* sPh = (uint16_t*)sPf;
            #pragma unroll
            for (int nt = 0; nt < 4; ++nt) {
                #pragma unroll
                for (int j = 0; j < 8; ++j) {
                    int r   = wave * 16 + j + 8 * hlf;
                    int col = nt * 16 + ln;
                    int pp  = col >> 1;
                    int idx = ((((pp >> 4) * 64) + r) * RSTR + (pp & 15)) * 2 + (col & 1);
                    sPh[idx] = (uint16_t)bfr_bits(accS[nt][j]);
                }
            }
        }
        __syncthreads();      // publish P (lane transpose)

        // ---- O += P(16x64) x V(64x256), 2-deep pipelined B fragments ----
        {
            v16bf avv[2];
            #pragma unroll
            for (int kc2 = 0; kc2 < 2; ++kc2) {
                const uint32_t* pa = sPf + ((kc2 * 64) + qrow) * RSTR + 4 * hlf;
                avv[kc2] = load_frag(pa, pa + 8);
            }
            auto loadVB = [&](int s) -> v16bf {
                int kc2 = s >> 4, ct = s & 15;
                const uint32_t* vb = sV + ((kc2 * 256) + ct * 16 + ln) * RSTR + 4 * hlf;
                return load_frag(vb, vb + 8);
            };
            v16bf bp[2];
            bp[0] = loadVB(0);
            bp[1] = loadVB(1);
            #pragma unroll
            for (int s = 0; s < 32; ++s) {
                v16bf curf = bp[s & 1];
                if (s + 2 < 32) bp[s & 1] = loadVB(s + 2);
                accO[s & 15] = __builtin_amdgcn_wmma_f32_16x16x32_bf16(
                    false, avv[s >> 4], false, curf, (short)0, accO[s & 15], false, false);
            }
        }

        wait_async0();        // next tile landed; everyone done with cur + sPf
        __syncthreads();
    }

    // ---- epilogue: O/l, out-proj + BN + residual, coalesced f32x4 ----
    float invl[8];
    #pragma unroll
    for (int j = 0; j < 8; ++j) invl[j] = 1.0f / l_r[j];

    const int nb = q0 + wave * 16 + 8 * hlf;
    #pragma unroll
    for (int ct = 0; ct < 16; ++ct) {
        int c = ct * 16 + ln;
        float sc = combS[c], bi = combB[c];
        const float* xr   = xb + (size_t)c * HW_DIM + nb;
        float*       orow = out + (size_t)b * C_DIM * HW_DIM + (size_t)c * HW_DIM + nb;
        float4 r0 = *(const float4*)(xr);
        float4 r1 = *(const float4*)(xr + 4);
        float4 o0, o1;
        o0.x = accO[ct][0] * invl[0] * sc + bi + r0.x;
        o0.y = accO[ct][1] * invl[1] * sc + bi + r0.y;
        o0.z = accO[ct][2] * invl[2] * sc + bi + r0.z;
        o0.w = accO[ct][3] * invl[3] * sc + bi + r0.w;
        o1.x = accO[ct][4] * invl[4] * sc + bi + r1.x;
        o1.y = accO[ct][5] * invl[5] * sc + bi + r1.y;
        o1.z = accO[ct][6] * invl[6] * sc + bi + r1.z;
        o1.w = accO[ct][7] * invl[7] * sc + bi + r1.w;
        *(float4*)(orow)     = o0;
        *(float4*)(orow + 4) = o1;
    }
}

// ===========================================================================
// Fallback: single fused kernel, used when ws is too small.
// ===========================================================================
#define SMEM_FUSED ((10240 + 10240 + 10240 + 2560 + 1536) * 4)

__global__ __launch_bounds__(128)
void nl_attn_fused_kernel(const float* __restrict__ x,
                          const float* __restrict__ w_g,  const float* __restrict__ b_g,
                          const float* __restrict__ w_th, const float* __restrict__ b_th,
                          const float* __restrict__ w_ph, const float* __restrict__ b_ph,
                          const float* __restrict__ w_o,  const float* __restrict__ b_o,
                          const float* __restrict__ bn_g, const float* __restrict__ bn_b,
                          const float* __restrict__ bn_m, const float* __restrict__ bn_v,
                          float* __restrict__ out)
{
    extern __shared__ unsigned char smem_raw[];
    uint32_t* sQf  = (uint32_t*)smem_raw;
    uint32_t* sKf  = sQf + 8 * 64 * RSTR;
    uint32_t* sVf  = sKf + 8 * 64 * RSTR;
    uint32_t* sPf  = sVf + 2 * 256 * RSTR;
    float*    combS = (float*)(sPf + 2 * 64 * RSTR);
    float*    combB = combS + 256;
    float*    sWph  = combB + 256;
    float*    sBph  = sWph + 256;
    float*    sWg   = sBph + 256;
    float*    sBg   = sWg + 256;

    const int b    = blockIdx.y;
    const int q0   = blockIdx.x * QT;
    const int tid  = threadIdx.x;
    const int lane = tid & 31;
    const int wave = tid >> 5;
    const int hlf  = lane >> 4;
    const int ln   = lane & 15;
    const float* xb = x + (size_t)b * C_DIM * HW_DIM;

    for (int c = tid; c < C_DIM; c += 128) {
        float iv = bn_g[c] * rsqrtf(bn_v[c] + 1e-5f);
        combS[c] = w_o[c] * iv;
        combB[c] = b_o[c] * iv + bn_b[c] - bn_m[c] * iv;
        sWph[c] = w_ph[c];  sBph[c] = b_ph[c];
        sWg[c]  = w_g[c];   sBg[c]  = b_g[c];
    }
    {
        const int n   = tid & 63;
        const int cpb = tid >> 6;
        #pragma unroll 4
        for (int it = 0; it < 64; ++it) {
            int cp = cpb + 2 * it;
            int c0 = 2 * cp;
            float v0 = xb[(size_t)c0 * HW_DIM + q0 + n]       * w_th[c0]     + b_th[c0];
            float v1 = xb[(size_t)(c0 + 1) * HW_DIM + q0 + n] * w_th[c0 + 1] + b_th[c0 + 1];
            sQf[(((cp >> 4) * 64) + n) * RSTR + (cp & 15)] = pack2bf(v0, v1);
        }
    }
    __syncthreads();

    const int qrow = wave * 16 + ln;
    v16bf aQ[8];
    #pragma unroll
    for (int kc = 0; kc < 8; ++kc) {
        const uint32_t* base = sQf + ((kc * 64) + qrow) * RSTR + 4 * hlf;
        aQ[kc] = load_frag(base, base + 8);
    }

    v8f accO[16];
    {
        v8f z = {0.f, 0.f, 0.f, 0.f, 0.f, 0.f, 0.f, 0.f};
        #pragma unroll
        for (int i = 0; i < 16; ++i) accO[i] = z;
    }
    float m_r[8], l_r[8];
    #pragma unroll
    for (int j = 0; j < 8; ++j) { m_r[j] = -1e30f; l_r[j] = 0.f; }

    for (int kb = 0; kb < NKB; ++kb) {
        const int k0g = kb * KT;
        __syncthreads();
        {
            const int key = tid & 63;
            const int cpb = tid >> 6;
            #pragma unroll 4
            for (int it = 0; it < 64; ++it) {
                int cp = cpb + 2 * it;
                int c0 = 2 * cp;
                float v0 = xb[(size_t)c0 * HW_DIM + k0g + key]       * sWph[c0]     + sBph[c0];
                float v1 = xb[(size_t)(c0 + 1) * HW_DIM + k0g + key] * sWph[c0 + 1] + sBph[c0 + 1];
                sKf[(((cp >> 4) * 64) + key) * RSTR + (cp & 15)] = pack2bf(v0, v1);
            }
        }
        {
            const int p  = tid & 31;
            const int cb = tid >> 5;
            const int kc2 = p >> 4, sv = p & 15;
            #pragma unroll 4
            for (int it = 0; it < 64; ++it) {
                int c = cb + 4 * it;
                float wc = sWg[c], bc = sBg[c];
                const float* xr = xb + (size_t)c * HW_DIM + k0g + 2 * p;
                sVf[((kc2 * 256) + c) * RSTR + sv] = pack2bf(xr[0] * wc + bc, xr[1] * wc + bc);
            }
        }
        __syncthreads();

        v8f accS[4];
        {
            v8f z = {0.f, 0.f, 0.f, 0.f, 0.f, 0.f, 0.f, 0.f};
            #pragma unroll
            for (int i = 0; i < 4; ++i) accS[i] = z;
        }
        #pragma unroll
        for (int kc = 0; kc < 8; ++kc) {
            #pragma unroll
            for (int nt = 0; nt < 4; ++nt) {
                const uint32_t* bb = sKf + ((kc * 64) + nt * 16 + ln) * RSTR + 4 * hlf;
                v16bf bv = load_frag(bb, bb + 8);
                accS[nt] = __builtin_amdgcn_wmma_f32_16x16x32_bf16(
                    false, aQ[kc], false, bv, (short)0, accS[nt], false, false);
            }
        }
        #pragma unroll
        for (int j = 0; j < 8; ++j) {
            float mv = fmaxf(fmaxf(accS[0][j], accS[1][j]),
                             fmaxf(accS[2][j], accS[3][j]));
            mv = fmaxf(mv, __shfl_xor(mv, 1, 16));
            mv = fmaxf(mv, __shfl_xor(mv, 2, 16));
            mv = fmaxf(mv, __shfl_xor(mv, 4, 16));
            mv = fmaxf(mv, __shfl_xor(mv, 8, 16));
            float mn = fmaxf(m_r[j], mv);
            float al = __expf(m_r[j] - mn);
            m_r[j] = mn;
            float rs = 0.f;
            #pragma unroll
            for (int nt = 0; nt < 4; ++nt) {
                float p = __expf(accS[nt][j] - mn);
                accS[nt][j] = p;
                rs += p;
            }
            rs += __shfl_xor(rs, 1, 16);
            rs += __shfl_xor(rs, 2, 16);
            rs += __shfl_xor(rs, 4, 16);
            rs += __shfl_xor(rs, 8, 16);
            l_r[j] = l_r[j] * al + rs;
            #pragma unroll
            for (int ct = 0; ct < 16; ++ct) accO[ct][j] *= al;
        }
        {
            uint16_t* sPh = (uint16_t*)sPf;
            #pragma unroll
            for (int nt = 0; nt < 4; ++nt) {
                #pragma unroll
                for (int j = 0; j < 8; ++j) {
                    int r   = wave * 16 + j + 8 * hlf;
                    int col = nt * 16 + ln;
                    int pp  = col >> 1;
                    int idx = ((((pp >> 4) * 64) + r) * RSTR + (pp & 15)) * 2 + (col & 1);
                    sPh[idx] = (uint16_t)bfr_bits(accS[nt][j]);
                }
            }
        }
        __syncthreads();

        #pragma unroll
        for (int kc2 = 0; kc2 < 2; ++kc2) {
            const uint32_t* pa = sPf + ((kc2 * 64) + qrow) * RSTR + 4 * hlf;
            v16bf av = load_frag(pa, pa + 8);
            #pragma unroll
            for (int ct = 0; ct < 16; ++ct) {
                const uint32_t* vb = sVf + ((kc2 * 256) + ct * 16 + ln) * RSTR + 4 * hlf;
                v16bf bv = load_frag(vb, vb + 8);
                accO[ct] = __builtin_amdgcn_wmma_f32_16x16x32_bf16(
                    false, av, false, bv, (short)0, accO[ct], false, false);
            }
        }
    }

    float invl[8];
    #pragma unroll
    for (int j = 0; j < 8; ++j) invl[j] = 1.0f / l_r[j];
    const int nb = q0 + wave * 16 + 8 * hlf;
    #pragma unroll
    for (int ct = 0; ct < 16; ++ct) {
        int c = ct * 16 + ln;
        float sc = combS[c], bi = combB[c];
        const float* xr   = xb + (size_t)c * HW_DIM + nb;
        float*       orow = out + (size_t)b * C_DIM * HW_DIM + (size_t)c * HW_DIM + nb;
        float4 r0 = *(const float4*)(xr);
        float4 r1 = *(const float4*)(xr + 4);
        float4 o0, o1;
        o0.x = accO[ct][0] * invl[0] * sc + bi + r0.x;
        o0.y = accO[ct][1] * invl[1] * sc + bi + r0.y;
        o0.z = accO[ct][2] * invl[2] * sc + bi + r0.z;
        o0.w = accO[ct][3] * invl[3] * sc + bi + r0.w;
        o1.x = accO[ct][4] * invl[4] * sc + bi + r1.x;
        o1.y = accO[ct][5] * invl[5] * sc + bi + r1.y;
        o1.z = accO[ct][6] * invl[6] * sc + bi + r1.z;
        o1.w = accO[ct][7] * invl[7] * sc + bi + r1.w;
        *(float4*)(orow)     = o0;
        *(float4*)(orow + 4) = o1;
    }
}

// ===========================================================================
extern "C" void kernel_launch(void* const* d_in, const int* in_sizes, int n_in,
                              void* d_out, int out_size, void* d_ws, size_t ws_size,
                              hipStream_t stream) {
    (void)in_sizes; (void)n_in; (void)out_size;
    const float* x    = (const float*)d_in[0];
    const float* w_g  = (const float*)d_in[1];
    const float* b_g  = (const float*)d_in[2];
    const float* w_th = (const float*)d_in[3];
    const float* b_th = (const float*)d_in[4];
    const float* w_ph = (const float*)d_in[5];
    const float* b_ph = (const float*)d_in[6];
    const float* w_o  = (const float*)d_in[7];
    const float* b_o  = (const float*)d_in[8];
    const float* bn_g = (const float*)d_in[9];
    const float* bn_b = (const float*)d_in[10];
    const float* bn_m = (const float*)d_in[11];
    const float* bn_v = (const float*)d_in[12];
    float* out = (float*)d_out;

    dim3 grid(HW_DIM / QT, 8, 1);
    dim3 block(128, 1, 1);

    if (ws_size >= (size_t)WS_BYTES_NEEDED) {
        uint32_t* Tg = (uint32_t*)d_ws;
        uint32_t* Pg = Tg + WS_U32_PER_ARR;
        uint32_t* Vg = Pg + WS_U32_PER_ARR;

        prep_theta_phi_kernel<<<WS_U32_PER_ARR / 256, 256, 0, stream>>>(
            x, w_th, b_th, w_ph, b_ph, Tg, Pg);
        prep_g_kernel<<<WS_U32_PER_ARR / 256, 256, 0, stream>>>(
            x, w_g, b_g, Vg);

        (void)hipFuncSetAttribute((const void*)nl_attn_async_kernel,
                                  hipFuncAttributeMaxDynamicSharedMemorySize,
                                  SMEM_ASYNC);
        nl_attn_async_kernel<<<grid, block, SMEM_ASYNC, stream>>>(
            x, Tg, Pg, Vg, w_o, b_o, bn_g, bn_b, bn_m, bn_v, out);
    } else {
        (void)hipFuncSetAttribute((const void*)nl_attn_fused_kernel,
                                  hipFuncAttributeMaxDynamicSharedMemorySize,
                                  SMEM_FUSED);
        nl_attn_fused_kernel<<<grid, block, SMEM_FUSED, stream>>>(
            x, w_g, b_g, w_th, b_th, w_ph, b_ph, w_o, b_o,
            bn_g, bn_b, bn_m, bn_v, out);
    }
}
